// RelativeAttn_11003706213220
// MI455X (gfx1250) — matrix operand order repaired and checked by reading, not compiled, not run on hardware
//
#include <hip/hip_runtime.h>

// ---------------------------------------------------------------------------
// Transformer-XL relative attention for MI455X (gfx1250, wave32, WMMA).
// B=4, S=2048, DIM=512, H=8, DH=64.
// All GEMM-like work runs on v_wmma_f32_16x16x32_f16 (f16 in, f32 accumulate).
// Flash-style attention (no S x S materialization); rel_shift is fused via two
// windowed WMMA GEMMs over contiguous pe rows + cross-lane selection.
// Workspace use: ~63 MB.
// ---------------------------------------------------------------------------

typedef _Float16 h16;
typedef __attribute__((ext_vector_type(16))) _Float16 v16h;
typedef __attribute__((ext_vector_type(8)))  _Float16 v8h;
typedef __attribute__((ext_vector_type(8)))  float    v8f;

#define B_  4
#define S_  2048
#define D_  512
#define H_  8
#define DH_ 64

__device__ __forceinline__ v8f wmma_f16(v16h a, v16h b, v8f c) {
  // (neg_a, A, neg_b, B, c_mod, C, reuse_a, reuse_b)
  return __builtin_amdgcn_wmma_f32_16x16x32_f16(false, a, false, b, (short)0, c,
                                                false, false);
}

// A-operand (16x32 f16, row-major source): lane m=l%16 is row M=m.
// Lanes 0-15 hold K=[0..7],[16..23]; lanes 16-31 hold K=[8..15],[24..31].
__device__ __forceinline__ v16h load_a_frag(const h16* A, long row0, int lda, int c0) {
  int l = threadIdx.x & 31;
  int m = l & 15;
  int kb = (l >> 4) << 3;
  const h16* p = A + (size_t)(row0 + m) * lda + c0 + kb;
  v8h lo = *(const v8h*)p;
  v8h hh = *(const v8h*)(p + 16);
  v16h r;
#pragma unroll
  for (int i = 0; i < 8; ++i) { r[i] = lo[i]; r[i + 8] = hh[i]; }
  return r;
}

__device__ __forceinline__ v16h load_a_frag_clamp(const h16* A, long row0, int lda,
                                                  int c0, long rmax) {
  int l = threadIdx.x & 31;
  int m = l & 15;
  int kb = (l >> 4) << 3;
  long row = row0 + m;
  if (row > rmax) row = rmax;
  const h16* p = A + (size_t)row * lda + c0 + kb;
  v8h lo = *(const v8h*)p;
  v8h hh = *(const v8h*)(p + 16);
  v16h r;
#pragma unroll
  for (int i = 0; i < 8; ++i) { r[i] = lo[i]; r[i + 8] = hh[i]; }
  return r;
}

// B-operand (32x16 f16): B[kd][n] = R[row0+n][c0+kd].  Lane n=l%16 is column n;
// lanes 0-15 hold K=[0..7],[16..23]; lanes 16-31 hold K=[8..15],[24..31].
// Per lane: two 8-half chunks from one row of R.
__device__ __forceinline__ v16h load_b_frag(const h16* R, long row0, int ldr, int c0) {
  int l = threadIdx.x & 31;
  int n = l & 15;
  int kb = (l >> 4) << 3;
  const h16* p = R + (size_t)(row0 + n) * ldr + c0 + kb;
  v8h lo = *(const v8h*)p;
  v8h hh = *(const v8h*)(p + 16);
  v16h r;
#pragma unroll
  for (int i = 0; i < 8; ++i) { r[i] = lo[i]; r[i + 8] = hh[i]; }
  return r;
}

__device__ __forceinline__ v16h load_b_frag_clamp(const h16* R, long row0, int ldr,
                                                  int c0, long rmax) {
  int l = threadIdx.x & 31;
  int n = l & 15;
  int kb = (l >> 4) << 3;
  long row = row0 + n;
  if (row < 0) row = 0;
  if (row > rmax) row = rmax;
  const h16* p = R + (size_t)row * ldr + c0 + kb;
  v8h lo = *(const v8h*)p;
  v8h hh = *(const v8h*)(p + 16);
  v16h r;
#pragma unroll
  for (int i = 0; i < 8; ++i) { r[i] = lo[i]; r[i + 8] = hh[i]; }
  return r;
}

// ---------------------------------------------------------------------------
// Elementwise helpers
// ---------------------------------------------------------------------------
__global__ void cvt_kernel(const float* __restrict__ src, h16* __restrict__ dst, int n) {
  int i = blockIdx.x * 256 + threadIdx.x;
  if (i < n) dst[i] = (h16)src[i];
}

__global__ void bias_prep_kernel(const float* __restrict__ bq, const float* __restrict__ u,
                                 const float* __restrict__ vb, float* __restrict__ bQU,
                                 float* __restrict__ bQV) {
  int i = threadIdx.x;  // 512 threads, u/v_bias are (H,DH) -> flat 512
  bQU[i] = bq[i] + u[i];
  bQV[i] = bq[i] + vb[i];
}

__global__ void sinus_kernel(h16* __restrict__ dst) {
  int i = blockIdx.x * 256 + threadIdx.x;  // over S_*D_
  if (i >= S_ * D_) return;
  int pos = i >> 9;
  int col = i & (D_ - 1);
  float e = (float)(col & ~1) / (float)D_;
  float ang = (float)pos * powf(10000.f, -e);
  float v = (col & 1) ? cosf(ang) : sinf(ang);
  dst[i] = (h16)v;
}

// LayerNorm over last dim (512), biased variance, eps=1e-5; outputs f16.
__global__ void __launch_bounds__(256) ln_kernel(const float* __restrict__ x,
                                                 const float* __restrict__ g,
                                                 const float* __restrict__ be,
                                                 h16* __restrict__ out) {
  int row = blockIdx.x;
  int tid = threadIdx.x;
  const float* xr = x + (size_t)row * D_;
  float a = xr[tid], c = xr[tid + 256];
  __shared__ float red[256];
  red[tid] = a + c;
  __syncthreads();
  for (int s = 128; s > 0; s >>= 1) {
    if (tid < s) red[tid] += red[tid + s];
    __syncthreads();
  }
  float mu = red[0] * (1.f / (float)D_);
  __syncthreads();
  float da = a - mu, dc = c - mu;
  red[tid] = da * da + dc * dc;
  __syncthreads();
  for (int s = 128; s > 0; s >>= 1) {
    if (tid < s) red[tid] += red[tid + s];
    __syncthreads();
  }
  float inv = rsqrtf(red[0] * (1.f / (float)D_) + 1e-5f);
  out[(size_t)row * D_ + tid]       = (h16)(da * inv * g[tid] + be[tid]);
  out[(size_t)row * D_ + tid + 256] = (h16)(dc * inv * g[tid + 256] + be[tid + 256]);
}

// v (B,S,H,DH) f16 -> vt (B,H,DH,S) f16 so AV B-fragments load contiguously.
__global__ void vtrans_kernel(const h16* __restrict__ v, h16* __restrict__ vt) {
  size_t i = (size_t)blockIdx.x * 256 + threadIdx.x;  // B_*H_*DH_*S_ = 4194304
  if (i >= (size_t)B_ * H_ * DH_ * S_) return;
  int s = (int)(i & (S_ - 1));
  int d = (int)((i >> 11) & (DH_ - 1));
  int h = (int)((i >> 17) & (H_ - 1));
  int b = (int)(i >> 20);
  vt[i] = v[(((size_t)b * S_ + s) << 9) + h * DH_ + d];
}

// ---------------------------------------------------------------------------
// Generic WMMA GEMM:  C[M,N] = A(MxK) @ B(NxK)^T + bias[N]
// Block = 128 threads (4 waves); block tile 64(M) x 64(N); each wave does 16x64.
// Both operands are K-contiguous -> clean fragment loads, no LDS needed.
// ---------------------------------------------------------------------------
template <bool OUT_F16>
__global__ void __launch_bounds__(128) gemm_kernel(const h16* __restrict__ A,
                                                   const h16* __restrict__ Bw,
                                                   const float* __restrict__ bias,
                                                   void* __restrict__ Cout,
                                                   int M, int N, int K) {
  int wave = threadIdx.x >> 5;
  long row0 = (long)blockIdx.x * 64 + wave * 16;
  int col0 = blockIdx.y * 64;
  v8f acc[4] = {};
  for (int k0 = 0; k0 < K; k0 += 32) {
    v16h a = load_a_frag(A, row0, K, k0);
#pragma unroll
    for (int j = 0; j < 4; ++j) {
      v16h b = load_b_frag(Bw, col0 + j * 16, K, k0);
      acc[j] = wmma_f16(a, b, acc[j]);
    }
  }
  int l = threadIdx.x & 31;
  int n = l & 15, hf = l >> 4;
#pragma unroll
  for (int j = 0; j < 4; ++j) {
    int col = col0 + j * 16 + n;
    float bb = bias ? bias[col] : 0.f;
#pragma unroll
    for (int r = 0; r < 8; ++r) {
      long row = row0 + r + hf * 8;
      float v = acc[j][r] + bb;
      if (OUT_F16) ((h16*)Cout)[row * (long)N + col] = (h16)v;
      else         ((float*)Cout)[row * (long)N + col] = v;
    }
  }
}

// ---------------------------------------------------------------------------
// Flash attention with fused Transformer-XL rel_shift.
// One wave per 16-row q-tile of one (b,h); 4 waves per block.
// Per 32-wide k-step: content (4 WMMA), pos windows W1/W2 (16 WMMA),
// online softmax, P restaged via LDS, AV (4 WMMA).
//
// rel_shift closed form (s1=s2=S):
//   shifted[q,k] = qv[q]   . pe[S-1-q+k]   if k <= q
//                = 0                        if k == q+1
//                = qv[q+1] . pe[k-q-2]      if k >  q+1
// Within a 16x16 tile both cases index contiguous 31-row pe windows:
//   W1[qi][t] = qv[q0+qi]   . pe[(S-16-q0+kk) + t]
//   W2[qi][t] = qv[q0+qi+1] . pe[(kk-q0-17)  + t],  t = 15 - qi + kj in [0,30]
// ---------------------------------------------------------------------------
__global__ void __launch_bounds__(128) flash_kernel(const h16* __restrict__ qu,
                                                    const h16* __restrict__ qv,
                                                    const h16* __restrict__ km,
                                                    const h16* __restrict__ vt,
                                                    const h16* __restrict__ pe,
                                                    const unsigned char* __restrict__ mask,
                                                    h16* __restrict__ aout) {
  int wave = threadIdx.x >> 5;
  int l = threadIdx.x & 31;
  int task = blockIdx.x * 4 + wave;        // B_*H_*(S_/16) = 4096 tasks
  int qt = task & (S_ / 16 - 1);           // & 127
  int bh = task >> 7;
  int b = bh >> 3, h = bh & (H_ - 1);
  int q0 = qt << 4;
  int n = l & 15, hf = l >> 4;
  long rowb = (long)b * S_;
  int col0 = h * DH_;

  v16h aqu[2], aqv[2], aqv1[2];
#pragma unroll
  for (int c = 0; c < 2; ++c) {
    aqu[c]  = load_a_frag(qu, rowb + q0, D_, col0 + c * 32);
    aqv[c]  = load_a_frag(qv, rowb + q0, D_, col0 + c * 32);
    aqv1[c] = load_a_frag_clamp(qv, rowb + q0 + 1, D_, col0 + c * 32, rowb + S_ - 1);
  }

  v8f O[4] = {};
  float m_r[8], l_r[8];
#pragma unroll
  for (int r = 0; r < 8; ++r) { m_r[r] = -1e30f; l_r[r] = 0.f; }

  __shared__ __align__(16) h16 Pb[4][16 * 40];  // P tile staging, ld=40 (padded)
  h16* P = Pb[wave];

  for (int k0 = 0; k0 < S_; k0 += 32) {
    float sc[2][8];
#pragma unroll
    for (int sub = 0; sub < 2; ++sub) {
      int kk = k0 + sub * 16;
      // content scores: (q+u) . k^T over DH=64
      v8f cacc = {};
#pragma unroll
      for (int c = 0; c < 2; ++c)
        cacc = wmma_f16(aqu[c], load_b_frag(km, rowb + kk, D_, col0 + c * 32), cacc);
      // pos windows
      int r1 = S_ - 16 - q0 + kk;
      int r2 = kk - q0 - 17;
      v8f w1a = {}, w1b = {}, w2a = {}, w2b = {};
#pragma unroll
      for (int c = 0; c < 2; ++c) {
        w1a = wmma_f16(aqv[c],  load_b_frag_clamp(pe, r1,      D_, col0 + c * 32, S_ - 1), w1a);
        w1b = wmma_f16(aqv[c],  load_b_frag_clamp(pe, r1 + 16, D_, col0 + c * 32, S_ - 1), w1b);
        w2a = wmma_f16(aqv1[c], load_b_frag_clamp(pe, r2,      D_, col0 + c * 32, S_ - 1), w2a);
        w2b = wmma_f16(aqv1[c], load_b_frag_clamp(pe, r2 + 16, D_, col0 + c * 32, S_ - 1), w2b);
      }
      // select shifted pos per element (C-layout: row M=r+8*hf, col N=n)
#pragma unroll
      for (int r = 0; r < 8; ++r) {
        int t = 15 - r - 8 * hf + n;            // 0..30, same for W1 and W2
        int src = (t & 15) | (hf << 4);         // source lane holding (M, t)
        float s1 = __shfl(w1a[r], src, 32);
        float s2 = __shfl(w1b[r], src, 32);
        float s3 = __shfl(w2a[r], src, 32);
        float s4 = __shfl(w2b[r], src, 32);
        int q = q0 + r + 8 * hf;
        int k = kk + n;
        float pos = (k <= q) ? (t < 16 ? s1 : s2)
                             : ((k == q + 1) ? 0.f : (t < 16 ? s3 : s4));
        float s = (cacc[r] + pos) * 0.044194173824159216f;  // 1/sqrt(512)
        if (mask[(rowb + q) * (long)S_ + k]) s = -1e9f;
        sc[sub][r] = s;
      }
    }
    // online softmax update (row reductions across the 16 lanes of each half)
#pragma unroll
    for (int r = 0; r < 8; ++r) {
      float mx = fmaxf(sc[0][r], sc[1][r]);
#pragma unroll
      for (int off = 1; off < 16; off <<= 1) mx = fmaxf(mx, __shfl_xor(mx, off, 32));
      float mn = fmaxf(m_r[r], mx);
      float alpha = __expf(m_r[r] - mn);
      float p0 = __expf(sc[0][r] - mn);
      float p1 = __expf(sc[1][r] - mn);
      float ps = p0 + p1;
#pragma unroll
      for (int off = 1; off < 16; off <<= 1) ps += __shfl_xor(ps, off, 32);
      l_r[r] = l_r[r] * alpha + ps;
      m_r[r] = mn;
#pragma unroll
      for (int a = 0; a < 4; ++a) O[a][r] *= alpha;
      int M = r + 8 * hf;
      P[M * 40 + n]      = (h16)p0;   // C-layout -> row-major 16x32 f16 tile
      P[M * 40 + 16 + n] = (h16)p1;
    }
    __syncthreads();
    // restage P as A-fragment and accumulate O += P @ V  (Vt rows are d)
    v16h ap;
    {
      const h16* pp = &P[n * 40 + hf * 8];
      v8h lo = *(const v8h*)pp;
      v8h hh = *(const v8h*)(pp + 16);
#pragma unroll
      for (int i = 0; i < 8; ++i) { ap[i] = lo[i]; ap[i + 8] = hh[i]; }
    }
#pragma unroll
    for (int a = 0; a < 4; ++a) {
      v16h bv = load_b_frag(vt, (long)(b * H_ + h) * DH_ + a * 16, S_, k0);
      O[a] = wmma_f16(ap, bv, O[a]);
    }
    __syncthreads();
  }
  // normalize and write attn output (f16, (B,S,H,DH) flattened as (B*S, 512))
#pragma unroll
  for (int a = 0; a < 4; ++a) {
#pragma unroll
    for (int r = 0; r < 8; ++r) {
      float o = O[a][r] / l_r[r];
      long srow = rowb + q0 + r + 8 * hf;
      aout[srow * D_ + col0 + a * 16 + n] = (h16)o;
    }
  }
}

// ---------------------------------------------------------------------------
extern "C" void kernel_launch(void* const* d_in, const int* in_sizes, int n_in,
                              void* d_out, int out_size, void* d_ws, size_t ws_size,
                              hipStream_t stream) {
  (void)in_sizes; (void)n_in; (void)out_size; (void)ws_size;
  const float* spec = (const float*)d_in[0];
  const unsigned char* mask = (const unsigned char*)d_in[1];  // jnp.bool_ -> 1 byte
  const float* ln_g = (const float*)d_in[2];
  const float* ln_b = (const float*)d_in[3];
  const float* Wq = (const float*)d_in[4];
  const float* bq = (const float*)d_in[5];
  const float* Wk = (const float*)d_in[6];
  const float* bk = (const float*)d_in[7];
  const float* Wv = (const float*)d_in[8];
  const float* bv = (const float*)d_in[9];
  const float* Wpos = (const float*)d_in[10];
  const float* u = (const float*)d_in[11];
  const float* vb = (const float*)d_in[12];
  const float* Wo = (const float*)d_in[13];
  const float* bo = (const float*)d_in[14];
  float* out = (float*)d_out;

  // Workspace carve-up (~63 MB total)
  char* w = (char*)d_ws;
  auto carve = [&](size_t bytes) {
    char* p = w;
    w += (bytes + 255) & ~(size_t)255;
    return p;
  };
  const size_t XSZ = (size_t)B_ * S_ * D_ * sizeof(h16);   // 8 MB
  const size_t WSZ = (size_t)D_ * D_ * sizeof(h16);        // 0.5 MB
  const size_t PSZ = (size_t)S_ * D_ * sizeof(h16);        // 2 MB
  h16* x_h    = (h16*)carve(XSZ);
  h16* Wq_h   = (h16*)carve(WSZ);
  h16* Wk_h   = (h16*)carve(WSZ);
  h16* Wv_h   = (h16*)carve(WSZ);
  h16* Wo_h   = (h16*)carve(WSZ);
  h16* Wpos_h = (h16*)carve(WSZ);
  h16* sin_h  = (h16*)carve(PSZ);
  h16* pe_h   = (h16*)carve(PSZ);
  h16* qu_h   = (h16*)carve(XSZ);
  h16* qv_h   = (h16*)carve(XSZ);
  h16* k_h    = (h16*)carve(XSZ);
  h16* v_h    = (h16*)carve(XSZ);
  h16* vt_h   = (h16*)carve(XSZ);
  h16* attn_h = (h16*)carve(XSZ);
  float* bQU  = (float*)carve(D_ * sizeof(float));
  float* bQV  = (float*)carve(D_ * sizeof(float));

  const int WN = D_ * D_;  // 262144
  cvt_kernel<<<WN / 256, 256, 0, stream>>>(Wq, Wq_h, WN);
  cvt_kernel<<<WN / 256, 256, 0, stream>>>(Wk, Wk_h, WN);
  cvt_kernel<<<WN / 256, 256, 0, stream>>>(Wv, Wv_h, WN);
  cvt_kernel<<<WN / 256, 256, 0, stream>>>(Wo, Wo_h, WN);
  cvt_kernel<<<WN / 256, 256, 0, stream>>>(Wpos, Wpos_h, WN);

  ln_kernel<<<B_ * S_, 256, 0, stream>>>(spec, ln_g, ln_b, x_h);
  sinus_kernel<<<(S_ * D_) / 256, 256, 0, stream>>>(sin_h);
  bias_prep_kernel<<<1, 512, 0, stream>>>(bq, u, vb, bQU, bQV);

  // pe = sinusoid @ Wpos^T  (f16 out)
  gemm_kernel<true><<<dim3(S_ / 64, D_ / 64), 128, 0, stream>>>(
      sin_h, Wpos_h, nullptr, pe_h, S_, D_, D_);
  // q+u, q+v_bias, k, v projections (f16 out, biases folded)
  gemm_kernel<true><<<dim3(B_ * S_ / 64, D_ / 64), 128, 0, stream>>>(
      x_h, Wq_h, bQU, qu_h, B_ * S_, D_, D_);
  gemm_kernel<true><<<dim3(B_ * S_ / 64, D_ / 64), 128, 0, stream>>>(
      x_h, Wq_h, bQV, qv_h, B_ * S_, D_, D_);
  gemm_kernel<true><<<dim3(B_ * S_ / 64, D_ / 64), 128, 0, stream>>>(
      x_h, Wk_h, bk, k_h, B_ * S_, D_, D_);
  gemm_kernel<true><<<dim3(B_ * S_ / 64, D_ / 64), 128, 0, stream>>>(
      x_h, Wv_h, bv, v_h, B_ * S_, D_, D_);

  vtrans_kernel<<<(B_ * H_ * DH_ * S_) / 256, 256, 0, stream>>>(v_h, vt_h);

  flash_kernel<<<(B_ * H_ * (S_ / 16)) / 4, 128, 0, stream>>>(
      qu_h, qv_h, k_h, vt_h, pe_h, mask, attn_h);

  // final projection: out = attn @ Wo^T + bo  (f32 out)
  gemm_kernel<false><<<dim3(B_ * S_ / 64, D_ / 64), 128, 0, stream>>>(
      attn_h, Wo_h, bo, out, B_ * S_, D_, D_);
}
